// lstm_36481452212720
// MI455X (gfx1250) — matrix operand order repaired and checked by reading, not compiled, not run on hardware
//
#include <hip/hip_runtime.h>
#include <hip/hip_bf16.h>
#include <math.h>

// ---------------------------------------------------------------------------
// 2-layer LSTM (B=512,T=1024,F=64,H=48) + attention head for MI455X (gfx1250)
//  - projections and the recurrent h@Whh^T run on v_wmma_f32_16x16x32_bf16
//  - cell state lives in VGPRs; h round-trips through 2KB of LDS per step
//  - explicit addrspace(1) pointers -> global_load/global_store (not flat)
//  - branchless activations: v_exp_f32 + raw v_rcp_f32 (denominator in [1,2])
// ---------------------------------------------------------------------------

typedef __attribute__((ext_vector_type(16))) __bf16 bf16x16;
typedef __attribute__((ext_vector_type(8)))  float  floatx8;
typedef __attribute__((ext_vector_type(4)))  float  float4v;

#define GLB __attribute__((address_space(1)))
typedef const GLB float* gcfp;
typedef GLB float*       gfp;

#define B_    512
#define T_    1024
#define F_    64
#define H_    48
#define G4_   192          // 4*H
#define BT_   (B_ * T_)
#define A1_   20
#define NCLS_ 2

__device__ __forceinline__ float rcp_fast(float x) {
    return __builtin_amdgcn_rcpf(x);                       // single v_rcp_f32
}
__device__ __forceinline__ float sigmoid_fast(float x) {
    return rcp_fast(1.0f + __expf(-x));                    // denom in [1, inf); fine
}
__device__ __forceinline__ float tanh_fast(float x) {
    float ax = __builtin_fabsf(x);
    float e  = __expf(-2.0f * ax);                         // e in (0,1] -> denom in (1,2]
    float t  = (1.0f - e) * rcp_fast(1.0f + e);
    return __builtin_copysignf(t, x);
}

__device__ __forceinline__ float4v ld4(const float* p) {
    return *(const float4v*)p;            // 16B-aligned by construction
}

// pack 8 floats (two float4) into fragment elements [base, base+8)
__device__ __forceinline__ void pack8(bf16x16& f, int base, float4v a, float4v b) {
#pragma unroll
    for (int i = 0; i < 4; ++i) {
        f[base + i]     = (__bf16)a[i];
        f[base + 4 + i] = (__bf16)b[i];
    }
}

// ---------------------------------------------------------------------------
// Kernel 1: hoisted input projection  XG[m,n] = X[m,:K] @ W[n,:K]^T + bih[n]+bhh[n]
// K is a template constant (64 or 48); pad checks fold away.
// block = (32,4)  grid = (BT/16, 3)  -> 12 N-tiles over 192 columns.
// ---------------------------------------------------------------------------
template <int K>
__global__ __launch_bounds__(128)
void proj16_kernel(const float* __restrict__ X, const float* __restrict__ W,
                   const float* __restrict__ bih, const float* __restrict__ bhh,
                   float* __restrict__ XG)
{
    const int lane  = threadIdx.x;
    const int ntile = blockIdx.y * 4 + threadIdx.y;   // 0..11
    const int mtile = blockIdx.x;
    const int m     = lane & 15;
    const int half  = lane >> 4;
    const int n     = ntile * 16 + m;                 // C/B column for this lane

    // ---- A fragments: row m; elem e <-> K = 8*half+e (e<8), 16+8*half+(e-8) ----
    const float* xrow = X + (size_t)(mtile * 16 + m) * K;
    float4v x0 = ld4(xrow +      8 * half);
    float4v x1 = ld4(xrow +      8 * half + 4);
    float4v x2 = ld4(xrow + 16 + 8 * half);
    float4v x3 = ld4(xrow + 16 + 8 * half + 4);
    float4v x4 = ld4(xrow + 32 + 8 * half);
    float4v x5 = ld4(xrow + 32 + 8 * half + 4);
    float4v x6 = {}, x7 = {};
    if (K == 64) {                                     // constexpr-folds
        x6 = ld4(xrow + 48 + 8 * half);
        x7 = ld4(xrow + 48 + 8 * half + 4);
    }
    bf16x16 a0, a1;
    pack8(a0, 0, x0, x1);  pack8(a0, 8, x2, x3);
    pack8(a1, 0, x4, x5);  pack8(a1, 8, x6, x7);

    // ---- B fragments: column n; elem e <-> K = 8*half+e (e<8), 16+8*half+(e-8) ----
    const float* wrow = W + (size_t)n * K;
    float4v w0 = ld4(wrow +      8 * half);
    float4v w1 = ld4(wrow +      8 * half + 4);
    float4v w2 = ld4(wrow + 16 + 8 * half);
    float4v w3 = ld4(wrow + 16 + 8 * half + 4);
    float4v w4 = {}, w5 = {}, w6 = {}, w7 = {};
    if (K == 64) {
        w4 = ld4(wrow + 32 + 8 * half);
        w5 = ld4(wrow + 32 + 8 * half + 4);
        w6 = ld4(wrow + 48 + 8 * half);
        w7 = ld4(wrow + 48 + 8 * half + 4);
    } else {                                           // K=48: K=32..47 in elems 0-7 of both halves
        w4 = ld4(wrow + 32 + 8 * half);
        w5 = ld4(wrow + 32 + 8 * half + 4);
    }
    bf16x16 b0, b1;
    pack8(b0, 0, w0, w1);  pack8(b0, 8, w2, w3);
    pack8(b1, 0, w4, w5);  pack8(b1, 8, w6, w7);

    floatx8 acc = {};
    acc = __builtin_amdgcn_wmma_f32_16x16x32_bf16(false, a0, false, b0, (short)0, acc, false, false);
    acc = __builtin_amdgcn_wmma_f32_16x16x32_bf16(false, a1, false, b1, (short)0, acc, false, false);

    const float bias = bih[n] + bhh[n];
    gfp orow = (gfp)(XG + (size_t)(mtile * 16 + 8 * half) * G4_ + n);
#pragma unroll
    for (int r = 0; r < 8; ++r)
        orow[(size_t)r * G4_] = acc[r] + bias;         // immediate-offset global store clause
}

// ---------------------------------------------------------------------------
// Kernel 2: recurrent scan. grid = B/16 blocks x 96 threads (3 waves).
// wave w owns hidden columns [16w,16w+16) for ALL FOUR gates, so the cell
// update is purely register-resident. Whh fragments preloaded (loop-invariant).
// Per-row xg / Hout base pointers hoisted as addrspace(1); per-step loads are
// immediate-offset global_load_b32 clauses.
// ---------------------------------------------------------------------------
__global__ __launch_bounds__(96)
void lstm_scan_kernel(const float* __restrict__ XG, const float* __restrict__ Whh,
                      float* __restrict__ Hout)
{
    __shared__ __bf16 h_lds[16][64];   // h_{t-1}, K-padded to 64 (cols 48..63 == 0)

    const int tid  = threadIdx.x;
    const int wv   = tid >> 5;         // 0..2
    const int lane = tid & 31;
    const int m    = lane & 15;
    const int half = lane >> 4;
    const int ncol = wv * 16 + m;      // hidden unit 0..47 owned by this lane column
    const int b0   = blockIdx.x * 16;  // batch-row tile base

    for (int i = tid; i < 16 * 64; i += 96) ((__bf16*)h_lds)[i] = (__bf16)0.0f;

    // ---- Whh B-fragments: 4 gates x 2 K-halves (K=48 padded to 64) ----
    bf16x16 Bf[4][2];
#pragma unroll
    for (int g = 0; g < 4; ++g) {
        const float* wrow = Whh + (size_t)(g * H_ + ncol) * H_;
        float4v w0 = ld4(wrow + 16 * half);
        float4v w1 = ld4(wrow + 16 * half + 4);
        float4v w2 = ld4(wrow + 16 * half + 8);
        float4v w3 = ld4(wrow + 16 * half + 12);
        float4v w4 = {}, w5 = {}, w6 = {}, w7 = {};
        if (half == 0) {                               // K=32..47 live only in lanes 0-15
            w4 = ld4(wrow + 32);
            w5 = ld4(wrow + 36);
            w6 = ld4(wrow + 40);
            w7 = ld4(wrow + 44);
        }
        pack8(Bf[g][0], 0, w0, w1);  pack8(Bf[g][0], 8, w2, w3);
        pack8(Bf[g][1], 0, w4, w5);  pack8(Bf[g][1], 8, w6, w7);
    }

    // hoisted per-row global base pointers (bump by one timestep each iteration)
    gcfp xgp[8];
    gfp  hop[8];
#pragma unroll
    for (int r = 0; r < 8; ++r) {
        const int brow = b0 + r + 8 * half;
        xgp[r] = (gcfp)(XG   + (size_t)brow * T_ * G4_ + ncol);
        hop[r] = (gfp)(Hout + (size_t)brow * T_ * H_  + ncol);
    }

    float cst[8];                       // cell state, register-resident
#pragma unroll
    for (int r = 0; r < 8; ++r) cst[r] = 0.0f;

    __syncthreads();

    for (int t = 0; t < T_; ++t) {
        // A fragments of h_{t-1} from LDS (one 32B run per fragment per lane, K = 16*half + e)
        bf16x16 a0, a1;
#pragma unroll
        for (int e = 0; e < 8; ++e) {
            a0[e]     = h_lds[m][16 * half + e];
            a0[8 + e] = h_lds[m][16 * half + 8 + e];
            a1[e]     = h_lds[m][32 + 16 * half + e];
            a1[8 + e] = h_lds[m][40 + 16 * half + e];   // half 1: zero-pad region
        }
        __syncthreads();   // all waves done reading h_lds before anyone rewrites it

        // Gate accumulators = hoisted input projection; immediate-offset clause
        floatx8 acc[4];
#pragma unroll
        for (int r = 0; r < 8; ++r) {
#pragma unroll
            for (int g = 0; g < 4; ++g)
                acc[g][r] = xgp[r][g * H_];            // +0 / +192B / +384B / +576B
        }
        __builtin_prefetch((const float*)xgp[0] + G4_, 0, 0);   // next t (global_prefetch_b8)

        // gates += h_{t-1} @ Whh^T : 8 WMMAs, 4-way ILP across gates
#pragma unroll
        for (int g = 0; g < 4; ++g) {
            acc[g] = __builtin_amdgcn_wmma_f32_16x16x32_bf16(false, a0, false, Bf[g][0],
                                                             (short)0, acc[g], false, false);
            acc[g] = __builtin_amdgcn_wmma_f32_16x16x32_bf16(false, a1, false, Bf[g][1],
                                                             (short)0, acc[g], false, false);
        }

        // cell update fully in registers (same slot == same (batch row, unit) for all gates)
#pragma unroll
        for (int r = 0; r < 8; ++r) {
            float ig = sigmoid_fast(acc[0][r]);
            float fg = sigmoid_fast(acc[1][r]);
            float gg = tanh_fast(acc[2][r]);
            float og = sigmoid_fast(acc[3][r]);
            float c  = fg * cst[r] + ig * gg;
            cst[r]   = c;
            float h  = og * tanh_fast(c);
            hop[r][0] = h;
            h_lds[r + 8 * half][ncol] = (__bf16)h;
        }
#pragma unroll
        for (int r = 0; r < 8; ++r) { xgp[r] += G4_; hop[r] += H_; }
        __syncthreads();   // h_t visible to all waves for the next step
    }
}

// ---------------------------------------------------------------------------
// Kernel 3: attention scores  scores[b,t] = tanh(h @ Ws1^T + bs1) @ Ws2^T + bs2
// One thread per (b,t); Ws1/Ws2 staged through LDS; branchless tanh.
// ---------------------------------------------------------------------------
__global__ __launch_bounds__(256)
void attn_score_kernel(const float* __restrict__ Hseq, const float* __restrict__ Ws1,
                       const float* __restrict__ bs1, const float* __restrict__ Ws2,
                       const float* __restrict__ bs2, float* __restrict__ scores)
{
    __shared__ float ws1[A1_ * H_];
    __shared__ float ws2[A1_];
    __shared__ float sb1[A1_];
    const int tid = threadIdx.x;
    for (int i = tid; i < A1_ * H_; i += 256) ws1[i] = Ws1[i];
    if (tid < A1_) { ws2[tid] = Ws2[tid]; sb1[tid] = bs1[tid]; }
    __syncthreads();

    const size_t idx = (size_t)blockIdx.x * 256 + tid;   // flat b*T+t
    gcfp h = (gcfp)(Hseq + idx * H_);
    float hr[H_];
#pragma unroll
    for (int j = 0; j < H_; ++j) hr[j] = h[j];

    float sc = bs2[0];
#pragma unroll 4
    for (int a = 0; a < A1_; ++a) {
        float d = sb1[a];
#pragma unroll
        for (int j = 0; j < H_; ++j) d += ws1[a * H_ + j] * hr[j];
        sc += tanh_fast(d) * ws2[a];
    }
    ((gfp)scores)[idx] = sc;
}

// ---------------------------------------------------------------------------
// Kernel 4: per-batch softmax over T, attention pooling, FCs, +stat.
// One block (256 threads) per batch row.
// ---------------------------------------------------------------------------
__global__ __launch_bounds__(256)
void attn_pool_fc_kernel(const float* __restrict__ Hseq, const float* __restrict__ scores,
                         const float* __restrict__ Wfc, const float* __restrict__ bfc,
                         const float* __restrict__ Wlab, const float* __restrict__ blab,
                         const float* __restrict__ stat, float* __restrict__ out)
{
    __shared__ float sc[T_];
    __shared__ float red[256 * H_];    // 48 KB partial-sum matrix
    __shared__ float rtmp[256];
    __shared__ float hid[H_];
    __shared__ float fcv[A1_];

    const int tid = threadIdx.x;
    const int b   = blockIdx.x;

    for (int i = tid; i < T_; i += 256) sc[i] = scores[(size_t)b * T_ + i];
    __syncthreads();

    // max over T
    float lm = -3.4e38f;
    for (int i = tid; i < T_; i += 256) lm = fmaxf(lm, sc[i]);
    rtmp[tid] = lm; __syncthreads();
    for (int s = 128; s > 0; s >>= 1) {
        if (tid < s) rtmp[tid] = fmaxf(rtmp[tid], rtmp[tid + s]);
        __syncthreads();
    }
    const float maxv = rtmp[0];
    __syncthreads();

    // sum of exp
    float ls = 0.0f;
    for (int i = tid; i < T_; i += 256) ls += __expf(sc[i] - maxv);
    rtmp[tid] = ls; __syncthreads();
    for (int s = 128; s > 0; s >>= 1) {
        if (tid < s) rtmp[tid] += rtmp[tid + s];
        __syncthreads();
    }
    const float inv = rcp_fast(rtmp[0]);   // sum >= 1 (includes max term)
    __syncthreads();

    // attention-weighted pooling: hidden[h] = sum_t attn[t] * H[b,t,h]
    float hacc[H_];
#pragma unroll
    for (int j = 0; j < H_; ++j) hacc[j] = 0.0f;
    for (int i = tid; i < T_; i += 256) {
        float w = __expf(sc[i] - maxv) * inv;
        gcfp hp = (gcfp)(Hseq + ((size_t)b * T_ + i) * H_);
#pragma unroll
        for (int j = 0; j < H_; ++j) hacc[j] += w * hp[j];
    }
#pragma unroll
    for (int j = 0; j < H_; ++j) red[tid * H_ + j] = hacc[j];
    __syncthreads();
    for (int s = 128; s > 0; s >>= 1) {
        if (tid < s) {
#pragma unroll
            for (int j = 0; j < H_; ++j) red[tid * H_ + j] += red[(tid + s) * H_ + j];
        }
        __syncthreads();
    }
    if (tid < H_) hid[tid] = red[tid];
    __syncthreads();

    // fc: H->20
    if (tid < A1_) {
        float a = bfc[tid];
#pragma unroll
        for (int j = 0; j < H_; ++j) a += Wfc[tid * H_ + j] * hid[j];
        fcv[tid] = a;
    }
    __syncthreads();

    // label: 20->2, + stat
    if (tid < NCLS_) {
        float a = blab[tid];
#pragma unroll
        for (int j = 0; j < A1_; ++j) a += Wlab[tid * A1_ + j] * fcv[j];
        out[(size_t)b * NCLS_ + tid] = a + stat[(size_t)b * NCLS_ + tid];
    }
}

// ---------------------------------------------------------------------------
// Host launcher
// ---------------------------------------------------------------------------
extern "C" void kernel_launch(void* const* d_in, const int* in_sizes, int n_in,
                              void* d_out, int out_size, void* d_ws, size_t ws_size,
                              hipStream_t stream)
{
    const float* x    = (const float*)d_in[0];
    const float* stat = (const float*)d_in[1];
    const float* Wih0 = (const float*)d_in[2];
    const float* Whh0 = (const float*)d_in[3];
    const float* bih0 = (const float*)d_in[4];
    const float* bhh0 = (const float*)d_in[5];
    const float* Wih1 = (const float*)d_in[6];
    const float* Whh1 = (const float*)d_in[7];
    const float* bih1 = (const float*)d_in[8];
    const float* bhh1 = (const float*)d_in[9];
    const float* Ws1  = (const float*)d_in[10];
    const float* bs1  = (const float*)d_in[11];
    const float* Ws2  = (const float*)d_in[12];
    const float* bs2  = (const float*)d_in[13];
    const float* Wfc  = (const float*)d_in[14];
    const float* bfc  = (const float*)d_in[15];
    const float* Wlab = (const float*)d_in[16];
    const float* blab = (const float*)d_in[17];

    // workspace layout (bytes)
    const size_t SZ_XG = (size_t)BT_ * G4_ * sizeof(float);   // 402.7 MB, reused per layer
    const size_t SZ_H  = (size_t)BT_ * H_  * sizeof(float);   // 100.7 MB
    char* ws = (char*)d_ws;
    float* xg     = (float*)(ws);
    float* h0     = (float*)(ws + SZ_XG);
    float* h1     = (float*)(ws + SZ_XG + SZ_H);
    float* scores = (float*)(ws + SZ_XG + 2 * SZ_H);
    float* outp   = (float*)d_out;

    dim3 pb(32, 4), pg(BT_ / 16, 3);

    // layer 0
    proj16_kernel<F_><<<pg, pb, 0, stream>>>(x, Wih0, bih0, bhh0, xg);
    lstm_scan_kernel<<<B_ / 16, 96, 0, stream>>>(xg, Whh0, h0);
    // layer 1 (xg buffer reused)
    proj16_kernel<H_><<<pg, pb, 0, stream>>>(h0, Wih1, bih1, bhh1, xg);
    lstm_scan_kernel<<<B_ / 16, 96, 0, stream>>>(xg, Whh1, h1);
    // attention head
    attn_score_kernel<<<BT_ / 256, 256, 0, stream>>>(h1, Ws1, bs1, Ws2, bs2, scores);
    attn_pool_fc_kernel<<<B_, 256, 0, stream>>>(h1, scores, Wfc, bfc, Wlab, blab, stat, outp);
}
